// HybridQLSTM_65481071398044
// MI455X (gfx1250) — matrix operand order repaired and checked by reading, not compiled, not run on hardware
//
#include <hip/hip_runtime.h>

typedef __attribute__((ext_vector_type(16))) _Float16 v16h;
typedef __attribute__((ext_vector_type(8)))  float    v8f;
typedef __attribute__((ext_vector_type(4)))  float    v4f;

#define FAN 68   // D(=64) + NQ(=4)

// ---------------------------------------------------------------------------
// Kernel 1: x-projection GEMM (S x 64) @ (64 x 16) via v_wmma_f32_16x16x32_f16.
// 16 output columns ordered gate-major: col = g*4 + q, g in {f,i,u,o}.
// Each wave owns 8 row-tiles of 16 rows (B-fragments built once, reused).
// Output kept in the NATIVE D-fragment lane layout:
//   pre[tile*256 + r*32 + lane]  <->  (row = tile*16 + r + (lane>>4)*8,
//                                      col = lane & 15)
// so every store is one contiguous 128B wave-store. Requires S % 16 == 0
// (holds: S = T*B with B a multiple of 16).
// ---------------------------------------------------------------------------
__global__ __launch_bounds__(256)
void qlstm_xproj_wmma(const float* __restrict__ X,
                      const float* __restrict__ Wf, const float* __restrict__ Wi,
                      const float* __restrict__ Wu, const float* __restrict__ Wo,
                      float* __restrict__ pre, unsigned S)
{
  const unsigned lane = threadIdx.x & 31u;
  const unsigned hi   = lane >> 4;       // half-wave select
  const unsigned nl   = lane & 15u;
  const unsigned wave = blockIdx.x * 8u + (threadIdx.x >> 5);

  // ---- B fragments: column nl = (gate g, qubit q); K = x-features.
  // Lanes 0-15 hold K=0..7,16..23 (chunk-relative), lanes 16-31 hold K=8..15,24..31.
  const unsigned g = nl >> 2, q = nl & 3u;
  const float* Wsel = (g == 0) ? Wf : (g == 1) ? Wi : (g == 2) ? Wu : Wo;
  const float* wrow = Wsel + q * FAN;          // first 64 entries = x-part
  v16h b0, b1;
#pragma unroll
  for (int e = 0; e < 16; ++e) {
    b0[e] = (_Float16)wrow[      (e >> 3) * 16 + hi * 8 + (e & 7)]; // K-chunk 0: K = 0..31
    b1[e] = (_Float16)wrow[32 +  (e >> 3) * 16 + hi * 8 + (e & 7)]; // K-chunk 1: K = 32..63
  }

  const unsigned tile0 = wave * 8u;
  const float* xr0 = X + (tile0 * 16u + nl) * 64u;   // A row: M = nl
  float*       pr0 = pre + tile0 * 256u + lane;

#pragma unroll 1
  for (unsigned s = 0; s < 8; ++s) {
    const unsigned tile = tile0 + s;
    if (tile * 16u >= S) break;                 // wave-uniform guard only
    const v4f* xr = (const v4f*)(xr0 + s * (16u * 64u));
    // A chunk0: elems 0-7 <-> K=hi*8+e ; elems 8-15 <-> K=16+hi*8+e
    v4f x0 = xr[     hi * 2 + 0], x1 = xr[     hi * 2 + 1];
    v4f x2 = xr[ 4 + hi * 2 + 0], x3 = xr[ 4 + hi * 2 + 1];
    v4f x4 = xr[ 8 + hi * 2 + 0], x5 = xr[ 8 + hi * 2 + 1];
    v4f x6 = xr[12 + hi * 2 + 0], x7 = xr[12 + hi * 2 + 1];
    v16h a0, a1;
#pragma unroll
    for (int e = 0; e < 4; ++e) {
      a0[e]      = (_Float16)x0[e];  a0[4 + e]  = (_Float16)x1[e];
      a0[8 + e]  = (_Float16)x2[e];  a0[12 + e] = (_Float16)x3[e];
      a1[e]      = (_Float16)x4[e];  a1[4 + e]  = (_Float16)x5[e];
      a1[8 + e]  = (_Float16)x6[e];  a1[12 + e] = (_Float16)x7[e];
    }
    v8f c = {};
    c = __builtin_amdgcn_wmma_f32_16x16x32_f16(false, a0, false, b0, (short)0, c, false, false);
    c = __builtin_amdgcn_wmma_f32_16x16x32_f16(false, a1, false, b1, (short)0, c, false, false);

    float* pr = pr0 + s * 256u;
#pragma unroll
    for (int r = 0; r < 8; ++r) pr[r * 32] = c[r];   // 8 x 128B wave-stores
  }
}

// ---------------------------------------------------------------------------
// Kernel 2: sequential LSTM recurrence, one thread per batch element.
// Reads `pre` through the native-fragment swizzle (still 64B contiguous/row).
// ---------------------------------------------------------------------------
__device__ __forceinline__ float sigm_f(float x) { return 1.0f / (1.0f + __expf(-x)); }
__device__ __forceinline__ float tanh_f(float x) { return 1.0f - 2.0f / (__expf(2.0f * x) + 1.0f); }

__global__ __launch_bounds__(256)
void qlstm_recur(const float* __restrict__ pre,
                 const float* __restrict__ Wf, const float* __restrict__ bf,
                 const float* __restrict__ Wi, const float* __restrict__ bi,
                 const float* __restrict__ Wu, const float* __restrict__ bu,
                 const float* __restrict__ Wo, const float* __restrict__ bo,
                 const float* __restrict__ thetas,
                 float* __restrict__ out, int T, int Bsz)
{
  __shared__ float sWh[16][4];   // recurrent weights W_g[q][64+j]
  __shared__ float sBT[16];      // bias + theta fused
  const int tid = threadIdx.x;
  if (tid < 16) {
    int g = tid >> 2, q = tid & 3;
    const float* Wsel = (g == 0) ? Wf : (g == 1) ? Wi : (g == 2) ? Wu : Wo;
    const float* bsel = (g == 0) ? bf : (g == 1) ? bi : (g == 2) ? bu : bo;
#pragma unroll
    for (int j = 0; j < 4; ++j) sWh[tid][j] = Wsel[q * FAN + 64 + j];
    sBT[tid] = bsel[q] + thetas[g * 4 + q];
  }
  __syncthreads();

  const int b = blockIdx.x * 256 + tid;
  if (b >= Bsz) return;

  // Swizzled base for row rr = t*Bsz + b in the native fragment layout:
  //   tile = t*(Bsz/16) + (b>>4), M = b&15
  //   offset = tile*256 + (M&7)*32 + (M>>3)*16  (+ col)
  const unsigned M = (unsigned)b & 15u;
  const float* pbase = pre + ((unsigned)(b >> 4) << 8) + ((M & 7u) << 5) + ((M >> 3) << 4);
  const size_t tstride = (size_t)Bsz * 16;       // floats per timestep

  float h[4]  = {0.f, 0.f, 0.f, 0.f};
  float cs[4] = {0.f, 0.f, 0.f, 0.f};

  for (int t = 0; t < T; ++t) {
    const float* p = pbase + (size_t)t * tstride;
    if (t + 1 < T) __builtin_prefetch(p + tstride, 0, 3);   // next step's row
    v4f r0 = ((const v4f*)p)[0], r1 = ((const v4f*)p)[1];
    v4f r2 = ((const v4f*)p)[2], r3 = ((const v4f*)p)[3];
    float a[16];
#pragma unroll
    for (int e = 0; e < 4; ++e) {
      a[e] = r0[e]; a[4 + e] = r1[e]; a[8 + e] = r2[e]; a[12 + e] = r3[e];
    }

    float z[16];
#pragma unroll
    for (int g = 0; g < 4; ++g) {
      float cg[4];
#pragma unroll
      for (int q = 0; q < 4; ++q) {
        const int k = g * 4 + q;
        float ang = a[k] + sBT[k]
                  + sWh[k][0] * h[0] + sWh[k][1] * h[1]
                  + sWh[k][2] * h[2] + sWh[k][3] * h[3];
        cg[q] = __cosf(ang);
      }
      // Closed-form PauliZ expectations after the CNOT ring:
      const float c23 = cg[2] * cg[3];
      z[g * 4 + 0] = cg[1] * c23;            // c1*c2*c3
      z[g * 4 + 1] = cg[0] * cg[1];          // c0*c1
      z[g * 4 + 2] = z[g * 4 + 1] * cg[2];   // c0*c1*c2
      z[g * 4 + 3] = z[g * 4 + 1] * c23;     // c0*c1*c2*c3
    }

    v4f hv;
#pragma unroll
    for (int q = 0; q < 4; ++q) {
      float fg = sigm_f(z[q]);
      float ig = sigm_f(z[4 + q]);
      float ug = tanh_f(z[8 + q]);
      float og = sigm_f(z[12 + q]);
      float cn = fg * cs[q] + ig * ug;
      cs[q] = cn;
      h[q]  = og * tanh_f(cn);
      hv[q] = h[q];
    }
    *(v4f*)(out + ((size_t)t * Bsz + b) * 4) = hv;   // stacked[t,b,:]
  }

  const size_t base = (size_t)T * Bsz * 4;
  v4f hv, cv;
#pragma unroll
  for (int q = 0; q < 4; ++q) { hv[q] = h[q]; cv[q] = cs[q]; }
  *(v4f*)(out + base + (size_t)b * 4) = hv;                       // hx
  *(v4f*)(out + base + (size_t)Bsz * 4 + (size_t)b * 4) = cv;     // cx
}

// ---------------------------------------------------------------------------
extern "C" void kernel_launch(void* const* d_in, const int* in_sizes, int n_in,
                              void* d_out, int out_size, void* d_ws, size_t ws_size,
                              hipStream_t stream) {
  const float* X  = (const float*)d_in[0];
  const float* Wf = (const float*)d_in[1];
  const float* bf = (const float*)d_in[2];
  const float* Wi = (const float*)d_in[3];
  const float* bi = (const float*)d_in[4];
  const float* Wu = (const float*)d_in[5];
  const float* bu = (const float*)d_in[6];
  const float* Wo = (const float*)d_in[7];
  const float* bo = (const float*)d_in[8];
  const float* th = (const float*)d_in[9];
  float* out = (float*)d_out;
  float* pre = (float*)d_ws;   // S*16 floats (33.5 MB for the reference sizes)

  const long long S   = (long long)in_sizes[0] / 64;        // T*B rows
  const long long Bsz = ((long long)out_size / 4 - S) / 2;  // out = 4*B*(T+2)
  const int T = (int)(S / Bsz);

  const long long tiles  = (S + 15) / 16;
  const long long waves  = (tiles + 7) / 8;   // 8 row-tiles per wave
  const int blocks1 = (int)((waves + 7) / 8); // 8 waves (256 thr) per block
  qlstm_xproj_wmma<<<blocks1, 256, 0, stream>>>(X, Wf, Wi, Wu, Wo, pre,
                                                (unsigned)S);

  const int blocks2 = (int)((Bsz + 255) / 256);
  qlstm_recur<<<blocks2, 256, 0, stream>>>(pre, Wf, bf, Wi, bi, Wu, bu,
                                           Wo, bo, th, out, T, (int)Bsz);
}
